// FilterDetections_1554778161626
// MI455X (gfx1250) — matrix operand order repaired and checked by reading, not compiled, not run on hardware
//
#include <hip/hip_runtime.h>

#define SCORE_THR 0.35f
#define IOU_THR   0.8f
#define POSE_HYPS 3
#define MAX_DET   300
#define NB 2
#define NA 1024
#define NC 8
#define PSTRIDE 1032   // halves per component row: 1032*2B=2064B -> bank offset 4*nn mod 64, conflict-free

typedef __attribute__((ext_vector_type(16))) _Float16 v16h;
typedef __attribute__((ext_vector_type(8)))  _Float16 v8h;
typedef __attribute__((ext_vector_type(8)))  float    v8f;

union V16U { v16h v; v8h h8[2]; };

// ---------------------------------------------------------------------------
// Phase 1: one workgroup per (batch, class). 1024 threads = 32 wave32 waves.
//  A) stage per-anchor box extents / conf / f16 poses (transposed+padded) in LDS
//  B) cluster[a] = first selected j with IoU(a,j) > 0.8 (self always hits)
//  C) per-row conf threshold + reciprocal of member count
//  D) pose averaging as W @ poses via v_wmma_f32_16x16x32_f16
// ---------------------------------------------------------------------------
__global__ __launch_bounds__(1024) void fd_phase1(
    const float* __restrict__ boxes, const float* __restrict__ cls,
    const float* __restrict__ poses, const float* __restrict__ conf,
    float* __restrict__ posework, int* __restrict__ denomw)
{
  __shared__ __align__(16) float4   s_box[NA];
  __shared__ __align__(16) float    s_conf[NA];
  __shared__ __align__(16) int      s_cluster[NA];
  __shared__ __align__(16) float    s_thr[NA];
  __shared__ __align__(16) float    s_inv[NA];      // 1/denom (0 if empty row)
  __shared__ __align__(16) int      s_tgt[NA];
  __shared__ __align__(16) _Float16 s_poseT[16 * PSTRIDE]; // [comp][anchor], comps 12..15 = 0
  __shared__ unsigned s_selmask[NA / 32];
  __shared__ int      s_numsel;

  const int bc  = blockIdx.x;
  const int b   = bc / NC, c = bc % NC;
  const int tid = threadIdx.x;

  if (tid < NA / 32) s_selmask[tid] = 0u;
  if (tid == 0)      s_numsel = 0;
  __syncthreads();

  // ---- Stage A ----
  {
    const float* bp = boxes + ((size_t)(b * NA + tid)) * (NC * 16) + c * 16;
    float xmn = bp[0], ymn = bp[1], xmx = bp[0], ymx = bp[1];
#pragma unroll
    for (int p = 1; p < 8; ++p) {
      float x = bp[2 * p], y = bp[2 * p + 1];
      xmn = fminf(xmn, x); xmx = fmaxf(xmx, x);
      ymn = fminf(ymn, y); ymx = fmaxf(ymx, y);
    }
    s_box[tid] = make_float4(xmn, ymn, xmx, ymx);
    float score = cls[(size_t)(b * NA + tid) * NC + c];
    if (score > SCORE_THR) {
      atomicOr(&s_selmask[tid >> 5], 1u << (tid & 31));
      atomicAdd(&s_numsel, 1);
    }
    s_conf[tid] = conf[(size_t)(b * NA + tid) * NC + c];
    const float* pp = poses + ((size_t)(b * NA + tid)) * (NC * 12) + c * 12;
#pragma unroll
    for (int k = 0; k < 16; ++k)
      s_poseT[k * PSTRIDE + tid] = (k < 12) ? (_Float16)pp[k] : (_Float16)0.0f;
  }
  __syncthreads();

  const bool selA    = ((s_selmask[tid >> 5] >> (tid & 31)) & 1u) != 0u;
  const int  repeats = min(POSE_HYPS, s_numsel);

  // ---- Stage B: cluster assignment; divide-free IoU test (un >= 1 always) ----
  int cl = -1;
  if (selA) {
    const float4 mb = s_box[tid];
    const float areaA = (mb.z - mb.x + 1.0f) * (mb.w - mb.y + 1.0f);
    for (int j = 0; j < NA; ++j) {
      if (!((s_selmask[j >> 5] >> (j & 31)) & 1u)) continue;
      float4 ob = s_box[j];
      float x1 = fmaxf(mb.x, ob.x), y1 = fmaxf(mb.y, ob.y);
      float x2 = fminf(mb.z, ob.z), y2 = fminf(mb.w, ob.w);
      float w = x2 - x1 + 1.0f, h = y2 - y1 + 1.0f;
      if (w > 0.0f && h > 0.0f) {
        float inter = w * h;
        float areaB = (ob.z - ob.x + 1.0f) * (ob.w - ob.y + 1.0f);
        float un = areaA + areaB - inter;
        if (un > 0.0f && inter > IOU_THR * un) { cl = j; break; }
      }
    }
  }
  s_cluster[tid] = cl;
  __syncthreads();

  // ---- Stage C: per-row threshold = k-th smallest conf among members ----
  {
    float c0 = 1e30f, c1 = 1e30f, c2 = 1e30f;
    int cnt = 0;
    if (selA) {  // only selected rows can be representatives
      const int4*   cl4 = (const int4*)s_cluster;
      const float4* cf4 = (const float4*)s_conf;
      for (int a4 = 0; a4 < NA / 4; ++a4) {
        int4   cc = cl4[a4];
        float4 vv = cf4[a4];
        int   cm[4] = {cc.x, cc.y, cc.z, cc.w};
        float cv[4] = {vv.x, vv.y, vv.z, vv.w};
#pragma unroll
        for (int q = 0; q < 4; ++q) {
          if (cm[q] == tid) {
            ++cnt;
            float v = cv[q];
            if      (v < c0) { c2 = c1; c1 = c0; c0 = v; }
            else if (v < c1) { c2 = c1; c1 = v; }
            else if (v < c2) { c2 = v; }
          }
        }
      }
    }
    int d = min(cnt, repeats);
    s_thr[tid] = (d == 0) ? -1e30f : (d == 1 ? c0 : (d == 2 ? c1 : c2));
    // denom in {0,1,2,3}: one reciprocal per row, exact constants
    s_inv[tid] = (d == 0) ? 0.0f : (d == 1 ? 1.0f : (d == 2 ? 0.5f : (1.0f / 3.0f)));
    denomw[(size_t)(b * NC + c) * NA + tid] = d;
  }
  __syncthreads();

  // column tid contributes to row cl iff its conf is within the row threshold
  s_tgt[tid] = (cl >= 0 && s_conf[tid] <= s_thr[cl]) ? cl : -1;
  __syncthreads();

  // ---- Stage D: D = W @ poses via WMMA f32<-f16, K swept over all 1024 ----
  {
    const int wave = tid >> 5;
    const int lane = tid & 31;
    const int g    = lane >> 4;   // lane half
    const int nn   = lane & 15;   // N (pose component) / M within tile
    const int m0a  = wave * 16;
    const int m0b  = (wave + 32) * 16;
    const _Float16* prow = &s_poseT[(size_t)nn * PSTRIDE];

    v8f acc0 = {};
    v8f acc1 = {};
    for (int s = 0; s < NA / 32; ++s) {
      const int k0 = s * 32;
      // B fragment: 16 contiguous padded halves -> two ds_load_b128, no predication
      const int kb = k0 + (g ? 16 : 0);
      V16U Bf;
      Bf.h8[0] = *(const v8h*)&prow[kb];
      Bf.h8[1] = *(const v8h*)&prow[kb + 8];
      // A fragment targets: 16 consecutive ints, same k order as B -> int4 loads
      const int kbA = k0 + (g ? 16 : 0);
      int4 t0 = *(const int4*)&s_tgt[kbA];
      int4 t1 = *(const int4*)&s_tgt[kbA + 4];
      int4 t2 = *(const int4*)&s_tgt[kbA + 8];
      int4 t3 = *(const int4*)&s_tgt[kbA + 12];
      int tg[16] = {t0.x, t0.y, t0.z, t0.w, t1.x, t1.y, t1.z, t1.w,
                    t2.x, t2.y, t2.z, t2.w, t3.x, t3.y, t3.z, t3.w};
      v16h Af0, Af1;
#pragma unroll
      for (int h = 0; h < 16; ++h) {
        Af0[h] = (tg[h] == m0a + nn) ? (_Float16)1.0f : (_Float16)0.0f;
        Af1[h] = (tg[h] == m0b + nn) ? (_Float16)1.0f : (_Float16)0.0f;
      }
      acc0 = __builtin_amdgcn_wmma_f32_16x16x32_f16(false, Af0, false, Bf.v,
                                                    (short)0, acc0, false, false);
      acc1 = __builtin_amdgcn_wmma_f32_16x16x32_f16(false, Af1, false, Bf.v,
                                                    (short)0, acc1, false, false);
    }
    // D layout: VGPR i -> row m0 + i (+8 for upper lane half), col = lane%16
    if (nn < 12) {
      float* base = posework + (size_t)(b * NC + c) * NA * 12 + nn;
#pragma unroll
      for (int i = 0; i < 8; ++i) {
        int r0 = m0a + i + (g ? 8 : 0);
        int r1 = m0b + i + (g ? 8 : 0);
        base[(size_t)r0 * 12] = acc0[i] * s_inv[r0];
        base[(size_t)r1 * 12] = acc1[i] * s_inv[r1];
      }
    }
  }
}

// ---------------------------------------------------------------------------
// Phase 2: one workgroup per batch. Top-300 sorted scores (bitonic in LDS)
// plus stable compaction of valid entries (block prefix scan).
// Output floats: scores[B*300] | labels[B*300] | poses[B*300*12] | idx[B*300]
// ---------------------------------------------------------------------------
__global__ __launch_bounds__(1024) void fd_phase2(
    const float* __restrict__ cls, const int* __restrict__ denomw,
    const float* __restrict__ posework, float* __restrict__ out)
{
  __shared__ float s_sc[NA * NC];
  __shared__ int   s_scan[1024];
  __shared__ int   s_nvalid;
  __shared__ int   s_base;

  const int b   = blockIdx.x;
  const int tid = threadIdx.x;
  float* scores_o = out + b * MAX_DET;
  float* labels_o = out + NB * MAX_DET + b * MAX_DET;
  float* poses_o  = out + 2 * NB * MAX_DET + b * MAX_DET * 12;
  float* idx_o    = out + 2 * NB * MAX_DET + NB * MAX_DET * 12 + b * MAX_DET;

  if (tid == 0) { s_nvalid = 0; s_base = 0; }
  __syncthreads();

  int lv = 0;
  for (int f = tid; f < NA * NC; f += 1024) {
    int c = f >> 10, a = f & (NA - 1);
    bool valid = denomw[(size_t)(b * NC + c) * NA + a] > 0;
    s_sc[f] = valid ? cls[(size_t)(b * NA + a) * NC + c] : -1e30f;
    lv += valid ? 1 : 0;
  }
  atomicAdd(&s_nvalid, lv);
  __syncthreads();
  const int nvalid = s_nvalid;

  // tail slots (slot >= nvalid) never touched by compaction: write -1 now
  for (int i = tid; i < MAX_DET; i += 1024) {
    if (i >= nvalid) {
      labels_o[i] = -1.0f;
      idx_o[i]    = -1.0f;
#pragma unroll
      for (int k = 0; k < 12; ++k) poses_o[i * 12 + k] = -1.0f;
    }
  }

  // bitonic sort, descending
  for (int k = 2; k <= NA * NC; k <<= 1) {
    for (int j = k >> 1; j > 0; j >>= 1) {
      __syncthreads();
      for (int i = tid; i < NA * NC; i += 1024) {
        int ixj = i ^ j;
        if (ixj > i) {
          float v0 = s_sc[i], v1 = s_sc[ixj];
          bool up = ((i & k) == 0);
          if (up ? (v0 < v1) : (v0 > v1)) { s_sc[i] = v1; s_sc[ixj] = v0; }
        }
      }
    }
  }
  __syncthreads();
  for (int i = tid; i < MAX_DET; i += 1024)
    scores_o[i] = (i < nvalid) ? s_sc[i] : -1.0f;

  // stable compaction of valid entries in flat (class-major) order
  for (int chunk = 0; chunk < (NA * NC) / 1024; ++chunk) {
    int f = chunk * 1024 + tid;
    int c = f >> 10, a = f & (NA - 1);
    int v = (denomw[(size_t)(b * NC + c) * NA + a] > 0) ? 1 : 0;
    s_scan[tid] = v;
    __syncthreads();
    for (int off = 1; off < 1024; off <<= 1) {
      int t = (tid >= off) ? s_scan[tid - off] : 0;
      __syncthreads();
      s_scan[tid] += t;
      __syncthreads();
    }
    int incl = s_scan[tid];
    int base = s_base;
    int slot = base + incl - v;
    if (v && slot < MAX_DET) {
      labels_o[slot] = (float)c;
      idx_o[slot]    = (float)a;
      const float* pw = posework + ((size_t)(b * NC + c) * NA + a) * 12;
#pragma unroll
      for (int k = 0; k < 12; ++k) poses_o[slot * 12 + k] = pw[k];
    }
    __syncthreads();
    if (tid == 1023) s_base = base + s_scan[1023];
    __syncthreads();
  }
}

extern "C" void kernel_launch(void* const* d_in, const int* in_sizes, int n_in,
                              void* d_out, int out_size, void* d_ws, size_t ws_size,
                              hipStream_t stream) {
  const float* boxes = (const float*)d_in[0];  // (B, A, C*16)
  const float* cls   = (const float*)d_in[1];  // (B, A, C)
  const float* poses = (const float*)d_in[2];  // (B, A, C*12)
  const float* conf  = (const float*)d_in[3];  // (B, A, C)

  float* posework = (float*)d_ws;                                 // B*C*A*12 floats
  int*   denomw   = (int*)(posework + (size_t)NB * NC * NA * 12); // B*C*A ints

  fd_phase1<<<NB * NC, 1024, 0, stream>>>(boxes, cls, poses, conf, posework, denomw);
  fd_phase2<<<NB, 1024, 0, stream>>>(cls, denomw, posework, (float*)d_out);
}
